// GAT_12532714569803
// MI455X (gfx1250) — matrix operand order repaired and checked by reading, not compiled, not run on hardware
//
#include <hip/hip_runtime.h>
#include <hip/hip_bf16.h>

typedef __attribute__((ext_vector_type(16))) _Float16 v16h;
typedef __attribute__((ext_vector_type(8)))  _Float16 v8h;
typedef __attribute__((ext_vector_type(8)))  float    v8f;

#define LEAKY(x) ((x) > 0.0f ? (x) : 0.2f * (x))
#define NEG_INF_ENC 0x007FFFFFu

__device__ __forceinline__ unsigned fenc(float f) {
    unsigned u = __float_as_uint(f);
    return (u & 0x80000000u) ? ~u : (u | 0x80000000u);
}
__device__ __forceinline__ float fdec(unsigned e) {
    return (e & 0x80000000u) ? __uint_as_float(e ^ 0x80000000u)
                             : __uint_as_float(~e);
}

// ---------- fill helpers ----------
__global__ void fill_f32(float* p, float v, long long n) {
    long long i = (long long)blockIdx.x * blockDim.x + threadIdx.x;
    long long st = (long long)gridDim.x * blockDim.x;
    for (; i < n; i += st) p[i] = v;
}
__global__ void fill_u32(unsigned* p, unsigned v, long long n) {
    long long i = (long long)blockIdx.x * blockDim.x + threadIdx.x;
    long long st = (long long)gridDim.x * blockDim.x;
    for (; i < n; i += st) p[i] = v;
}

// ---------- K0a: fuse rank-1 chains into small vectors ----------
// fused layout (floats): PL[64] QL[64] PR[64] QR[64] U1[64] V1[64] U2[128] V2[128]
__global__ void fuse_small(const float* __restrict__ W_em, const float* __restrict__ b_em,
                           const float* __restrict__ W0,   const float* __restrict__ b0,
                           const float* __restrict__ Wl1,  const float* __restrict__ bl1,
                           const float* __restrict__ Wr1,  const float* __restrict__ br1,
                           const float* __restrict__ We1,  const float* __restrict__ We2,
                           float* __restrict__ fused) {
    int t = threadIdx.x;
    if (t < 384) {
        int which = t / 64, c = t % 64;
        float s = 0.0f;
        if (which == 0) { for (int k = 0; k < 128; ++k) s += W0[k]  * Wl1[k * 64 + c]; }
        else if (which == 1) { for (int k = 0; k < 128; ++k) s += b0[k] * Wl1[k * 64 + c]; s += bl1[c]; }
        else if (which == 2) { for (int k = 0; k < 128; ++k) s += W0[k]  * Wr1[k * 64 + c]; }
        else if (which == 3) { for (int k = 0; k < 128; ++k) s += b0[k] * Wr1[k * 64 + c]; s += br1[c]; }
        else if (which == 4) { for (int k = 0; k < 128; ++k) s += W_em[k] * We1[k * 64 + c]; }
        else                 { for (int k = 0; k < 128; ++k) s += b_em[k] * We1[k * 64 + c]; }
        fused[which * 64 + c] = s;
    } else if (t < 640) {
        int u = t - 384;           // 0..255
        int which = u / 128, c = u % 128;
        float s = 0.0f;
        if (which == 0) { for (int k = 0; k < 128; ++k) s += W_em[k] * We2[k * 128 + c]; }
        else            { for (int k = 0; k < 128; ++k) s += b_em[k] * We2[k * 128 + c]; }
        fused[384 + which * 128 + c] = s;
    }
}

// ---------- K0b: convert + pack layer-2 weights into WMMA B-fragment order ----------
// packed[((colTile*2 + kchunk)*32 + lane)*16 + j]  with j=2v+odd ->
//   B[kchunk*32 + (j>>3)*16 + (lane>>4)*8 + (j&7)][colTile*16 + (lane&15)]
// so each lane's 16-half B fragment is one aligned 32-byte load in the GEMM.
__global__ void pack_w2(const float* __restrict__ Wl2, const float* __restrict__ Wr2,
                        _Float16* __restrict__ pl, _Float16* __restrict__ pr) {
    int i = blockIdx.x * blockDim.x + threadIdx.x;   // 0 .. 8191
    if (i >= 8192) return;
    int j      = i & 15;
    int lane   = (i >> 4) & 31;
    int cthunk = i >> 9;              // colTile*2 + kchunk
    int kchunk = cthunk & 1, ct = cthunk >> 1;
    int v = j >> 1, odd = j & 1;
    int half = lane >> 4, l = lane & 15;
    int col = ct * 16 + l;
    int kB  = kchunk * 32 + (j >> 3) * 16 + half * 8 + (j & 7);
    pl[i] = (_Float16)Wl2[kB * 128 + col];
    pr[i] = (_Float16)Wr2[kB * 128 + col];
}

// ---------- K1: layer-1 raw scores + segment max ----------
__global__ void edge_score1(const long long* __restrict__ ei, const float* __restrict__ x,
                            const float* __restrict__ ea, const float* __restrict__ fused,
                            const float* __restrict__ att1,
                            float* __restrict__ alpha1, unsigned* __restrict__ nmax1,
                            int E) {
    int idx = blockIdx.x * blockDim.x + threadIdx.x;
    if (idx >= E * 8) return;
    int e = idx >> 3, h = idx & 7;
    int s = (int)ei[e], d = (int)ei[E + e];
    float xs = x[s], xd = x[d], eav = ea[e];
    const float* PL = fused;       const float* QL = fused + 64;
    const float* PR = fused + 128; const float* QR = fused + 192;
    const float* U1 = fused + 256; const float* V1 = fused + 320;
    float sum = 0.0f;
    #pragma unroll
    for (int c = 0; c < 8; ++c) {
        int hc = h * 8 + c;
        float v = xs * PL[hc] + QL[hc] + xd * PR[hc] + QR[hc] + eav * U1[hc] + V1[hc];
        sum += LEAKY(v) * att1[hc];
    }
    alpha1[idx] = sum;
    atomicMax(&nmax1[d * 8 + h], fenc(sum));
}

// ---------- K2: exp + segment denom ----------
__global__ void edge_exp1(const long long* __restrict__ ei, float* __restrict__ alpha1,
                          const unsigned* __restrict__ nmax1, float* __restrict__ denom1,
                          int E) {
    int idx = blockIdx.x * blockDim.x + threadIdx.x;
    if (idx >= E * 8) return;
    int e = idx >> 3, h = idx & 7;
    int d = (int)ei[E + e];
    unsigned m = nmax1[d * 8 + h];
    float amax = (m == NEG_INF_ENC) ? 0.0f : fdec(m);
    float ex = expf(alpha1[idx] - amax);
    alpha1[idx] = ex;
    atomicAdd(&denom1[d * 8 + h], ex);
}

// ---------- K3: normalized weighted scatter of x[src] (rank-1 collapse) ----------
__global__ void edge_agg1(const long long* __restrict__ ei, const float* __restrict__ x,
                          const float* __restrict__ alpha1, const float* __restrict__ denom1,
                          float* __restrict__ S1, int E) {
    int idx = blockIdx.x * blockDim.x + threadIdx.x;
    if (idx >= E * 8) return;
    int e = idx >> 3, h = idx & 7;
    int s = (int)ei[e], d = (int)ei[E + e];
    float w = alpha1[idx] / fmaxf(denom1[d * 8 + h], 1e-16f);
    atomicAdd(&S1[d * 8 + h], w * x[s]);
}

// ---------- K4: reconstruct h1 = elu(out1 + bias1), store f16 ----------
__global__ void node_h1(const float* __restrict__ S1, const float* __restrict__ denom1,
                        const float* __restrict__ fused, const float* __restrict__ bias1,
                        _Float16* __restrict__ h1h, int N) {
    int idx = blockIdx.x * blockDim.x + threadIdx.x;
    if (idx >= N * 64) return;
    int n = idx >> 6, hc = idx & 63, h = hc >> 3;
    const float* PL = fused; const float* QL = fused + 64;
    float nz = denom1[n * 8 + h] > 0.0f ? 1.0f : 0.0f;
    float v = PL[hc] * S1[n * 8 + h] + QL[hc] * nz + bias1[hc];
    v = v > 0.0f ? v : (expf(v) - 1.0f);
    h1h[n * 64 + hc] = (_Float16)v;
}

// ---------- K5: WMMA GEMM  xl2/xr2 = h1[NPAD,64] @ W[64,128] + b ----------
// A fragments: two contiguous 16B runs per lane (b128 loads).
// B fragments: pre-packed, one aligned 32B run per lane (2x b128).
// XL/XR have NPAD rows, so stores are unguarded (padding rows never read).
__global__ void gemm2_wmma(const _Float16* __restrict__ A, const _Float16* __restrict__ BLp,
                           const _Float16* __restrict__ BRp, const float* __restrict__ bl2,
                           const float* __restrict__ br2, float* __restrict__ XL,
                           float* __restrict__ XR, int nRowTiles) {
    int wave = (blockIdx.x * blockDim.x + threadIdx.x) >> 5;
    int lane = threadIdx.x & 31;
    if (wave >= nRowTiles * 8) return;
    int rowTile = wave >> 3, colTile = wave & 7;
    int half = lane >> 4, l = lane & 15;
    int mrow = rowTile * 16 + l;
    int col = colTile * 16 + l;
    const _Float16* arow = A + mrow * 64;
    float blv = bl2[col];
    float brv = br2[col];
    v8f cL = {}; v8f cR = {};
    #pragma unroll
    for (int kcIdx = 0; kcIdx < 2; ++kcIdx) {
        int kbase = kcIdx * 32 + half * 8;
        v8h alo = *(const v8h*)(arow + kbase);         // K kbase .. kbase+7
        v8h ahi = *(const v8h*)(arow + kbase + 16);    // K kbase+16 .. kbase+23
        v16h a;
        #pragma unroll
        for (int j = 0; j < 8; ++j) { a[j] = alo[j]; a[8 + j] = ahi[j]; }
        const _Float16* fb = BLp + ((colTile * 2 + kcIdx) * 32 + lane) * 16;
        const _Float16* gb = BRp + ((colTile * 2 + kcIdx) * 32 + lane) * 16;
        v16h bl = *(const v16h*)fb;
        v16h br = *(const v16h*)gb;
        cL = __builtin_amdgcn_wmma_f32_16x16x32_f16(false, a, false, bl, (short)0, cL, false, false);
        cR = __builtin_amdgcn_wmma_f32_16x16x32_f16(false, a, false, br, (short)0, cR, false, false);
    }
    int rowBase = rowTile * 16 + half * 8;   // C/D: M = v + 8*(lane/16), N = lane%16
    #pragma unroll
    for (int v = 0; v < 8; ++v) {
        XL[(rowBase + v) * 128 + col] = cL[v] + blv;
        XR[(rowBase + v) * 128 + col] = cR[v] + brv;
    }
}

// ---------- K6: layer-2 raw score (one wave per edge) ----------
__global__ void edge_score2(const long long* __restrict__ ei, const float* __restrict__ ea,
                            const float* __restrict__ XL, const float* __restrict__ XR,
                            const float* __restrict__ fused, const float* __restrict__ att2,
                            float* __restrict__ alpha2, unsigned* __restrict__ nmax2, int E) {
    int e = (blockIdx.x * blockDim.x + threadIdx.x) >> 5;
    int lane = threadIdx.x & 31;
    if (e >= E) return;
    int s = (int)ei[e], d = (int)ei[E + e];
    float eav = ea[e];
    const float* U2 = fused + 384; const float* V2 = fused + 512;
    float sum = 0.0f;
    #pragma unroll
    for (int j = 0; j < 4; ++j) {
        int c = lane + 32 * j;
        float v = XL[(long long)s * 128 + c] + XR[(long long)d * 128 + c] + eav * U2[c] + V2[c];
        sum += LEAKY(v) * att2[c];
    }
    #pragma unroll
    for (int off = 16; off > 0; off >>= 1) sum += __shfl_xor(sum, off, 32);
    if (lane == 0) {
        alpha2[e] = sum;
        atomicMax(&nmax2[d], fenc(sum));
    }
}

// ---------- K7: exp + denom (head=1) ----------
__global__ void edge_exp2(const long long* __restrict__ ei, float* __restrict__ alpha2,
                          const unsigned* __restrict__ nmax2, float* __restrict__ denom2, int E) {
    int e = blockIdx.x * blockDim.x + threadIdx.x;
    if (e >= E) return;
    int d = (int)ei[E + e];
    unsigned m = nmax2[d];
    float amax = (m == NEG_INF_ENC) ? 0.0f : fdec(m);
    float ex = expf(alpha2[e] - amax);
    alpha2[e] = ex;
    atomicAdd(&denom2[d], ex);
}

// ---------- K8: weighted scatter of xl2[src] (one wave per edge) ----------
__global__ void edge_agg2(const long long* __restrict__ ei, const float* __restrict__ alpha2,
                          const float* __restrict__ denom2, const float* __restrict__ XL,
                          float* __restrict__ out2, int E) {
    int e = (blockIdx.x * blockDim.x + threadIdx.x) >> 5;
    int lane = threadIdx.x & 31;
    if (e >= E) return;
    int s = (int)ei[e], d = (int)ei[E + e];
    float w = alpha2[e] / fmaxf(denom2[d], 1e-16f);
    #pragma unroll
    for (int j = 0; j < 4; ++j) {
        int c = lane + 32 * j;
        atomicAdd(&out2[(long long)d * 128 + c], w * XL[(long long)s * 128 + c]);
    }
}

// ---------- K9: decoder pre-dot per node (one wave per node) ----------
__global__ void node_dec(const float* __restrict__ out2, const float* __restrict__ bias2,
                         const float* __restrict__ Wd, float* __restrict__ sa,
                         float* __restrict__ sb, int N) {
    int n = (blockIdx.x * blockDim.x + threadIdx.x) >> 5;
    int lane = threadIdx.x & 31;
    if (n >= N) return;
    float a = 0.0f, b = 0.0f;
    #pragma unroll
    for (int j = 0; j < 4; ++j) {
        int c = lane + 32 * j;
        float h = out2[(long long)n * 128 + c] + bias2[c];
        a += h * Wd[c];
        b += h * Wd[128 + c];
    }
    #pragma unroll
    for (int off = 16; off > 0; off >>= 1) {
        a += __shfl_xor(a, off, 32);
        b += __shfl_xor(b, off, 32);
    }
    if (lane == 0) { sa[n] = a; sb[n] = b; }
}

// ---------- K10: final per-edge output ----------
__global__ void edge_out(const long long* __restrict__ ei, const float* __restrict__ sa,
                         const float* __restrict__ sb, const float* __restrict__ bd,
                         float* __restrict__ out, int E) {
    int e = blockIdx.x * blockDim.x + threadIdx.x;
    if (e >= E) return;
    out[e] = sa[(int)ei[e]] + sb[(int)ei[E + e]] + bd[0];
}

extern "C" void kernel_launch(void* const* d_in, const int* in_sizes, int n_in,
                              void* d_out, int out_size, void* d_ws, size_t ws_size,
                              hipStream_t stream) {
    const float*     x    = (const float*)d_in[0];
    const long long* ei   = (const long long*)d_in[1];
    const float*     ea   = (const float*)d_in[2];
    const float*     W_em = (const float*)d_in[3];
    const float*     b_em = (const float*)d_in[4];
    const float*     W0   = (const float*)d_in[5];
    const float*     b0   = (const float*)d_in[6];
    const float*     Wl1  = (const float*)d_in[7];
    const float*     bl1  = (const float*)d_in[8];
    const float*     Wr1  = (const float*)d_in[9];
    const float*     br1  = (const float*)d_in[10];
    const float*     We1  = (const float*)d_in[11];
    const float*     att1 = (const float*)d_in[12];
    const float*     bias1= (const float*)d_in[13];
    const float*     Wl2  = (const float*)d_in[14];
    const float*     bl2  = (const float*)d_in[15];
    const float*     Wr2  = (const float*)d_in[16];
    const float*     br2  = (const float*)d_in[17];
    // We2 = d_in[18] consumed in fuse_small
    const float*     att2 = (const float*)d_in[19];
    const float*     bias2= (const float*)d_in[20];
    const float*     Wd   = (const float*)d_in[21];
    const float*     bd   = (const float*)d_in[22];
    float* out = (float*)d_out;

    const int N = in_sizes[0];          // 50000
    const int E = in_sizes[2];          // 800000
    const int NPAD = (N + 15) & ~15;

    size_t off = 0;
    auto carve = [&](size_t bytes) -> char* {
        char* p = (char*)d_ws + off;
        off = (off + bytes + 255) & ~(size_t)255;
        return p;
    };
    float*     fused  = (float*)carve(640 * sizeof(float));
    _Float16*  wl2p   = (_Float16*)carve(8192 * sizeof(_Float16));
    _Float16*  wr2p   = (_Float16*)carve(8192 * sizeof(_Float16));
    _Float16*  h1h    = (_Float16*)carve((size_t)NPAD * 64 * sizeof(_Float16));
    float*     XL     = (float*)carve((size_t)NPAD * 128 * sizeof(float));
    float*     XR     = (float*)carve((size_t)NPAD * 128 * sizeof(float));
    float*     out2   = (float*)carve((size_t)N * 128 * sizeof(float));
    float*     alpha1 = (float*)carve((size_t)E * 8 * sizeof(float));
    float*     alpha2 = (float*)carve((size_t)E * sizeof(float));
    unsigned*  nmax1  = (unsigned*)carve((size_t)N * 8 * sizeof(unsigned));
    float*     denom1 = (float*)carve((size_t)N * 8 * sizeof(float));
    float*     S1     = (float*)carve((size_t)N * 8 * sizeof(float));
    unsigned*  nmax2  = (unsigned*)carve((size_t)N * sizeof(unsigned));
    float*     denom2 = (float*)carve((size_t)N * sizeof(float));
    float*     sa     = (float*)carve((size_t)N * sizeof(float));
    float*     sb     = (float*)carve((size_t)N * sizeof(float));
    (void)ws_size; (void)n_in; (void)out_size;

    const int TB = 256;
    const int FB = 1024;   // blocks for grid-stride fills

    // reset accumulators (graph is replayed; atomics must start from scratch)
    fill_u32<<<FB, TB, 0, stream>>>((unsigned*)h1h, 0u, (long long)NPAD * 32);
    fill_f32<<<FB, TB, 0, stream>>>(out2, 0.0f, (long long)N * 128);
    fill_f32<<<FB, TB, 0, stream>>>(denom1, 0.0f, (long long)N * 8);
    fill_f32<<<FB, TB, 0, stream>>>(S1, 0.0f, (long long)N * 8);
    fill_f32<<<FB, TB, 0, stream>>>(denom2, 0.0f, (long long)N);
    fill_u32<<<FB, TB, 0, stream>>>(nmax1, NEG_INF_ENC, (long long)N * 8);
    fill_u32<<<FB, TB, 0, stream>>>(nmax2, NEG_INF_ENC, (long long)N);

    fuse_small<<<1, 1024, 0, stream>>>(W_em, b_em, W0, b0, Wl1, bl1, Wr1, br1,
                                       We1, (const float*)d_in[18], fused);
    pack_w2<<<32, TB, 0, stream>>>(Wl2, Wr2, wl2p, wr2p);

    int g_eh = (E * 8 + TB - 1) / TB;
    edge_score1<<<g_eh, TB, 0, stream>>>(ei, x, ea, fused, att1, alpha1, nmax1, E);
    edge_exp1  <<<g_eh, TB, 0, stream>>>(ei, alpha1, nmax1, denom1, E);
    edge_agg1  <<<g_eh, TB, 0, stream>>>(ei, x, alpha1, denom1, S1, E);

    node_h1<<<(N * 64 + TB - 1) / TB, TB, 0, stream>>>(S1, denom1, fused, bias1, h1h, N);

    int nRowTiles = NPAD / 16;
    int tiles = nRowTiles * 8;                 // 16x16 output tiles, 8 col tiles
    gemm2_wmma<<<(tiles + 7) / 8, TB, 0, stream>>>(h1h, wl2p, wr2p, bl2, br2, XL, XR,
                                                   nRowTiles);

    edge_score2<<<(E + 7) / 8, TB, 0, stream>>>(ei, ea, XL, XR, fused, att2, alpha2, nmax2, E);
    edge_exp2  <<<(E + TB - 1) / TB, TB, 0, stream>>>(ei, alpha2, nmax2, denom2, E);
    edge_agg2  <<<(E + 7) / 8, TB, 0, stream>>>(ei, alpha2, denom2, XL, out2, E);

    node_dec<<<(N + 7) / 8, TB, 0, stream>>>(out2, bias2, Wd, sa, sb, N);
    edge_out<<<(E + TB - 1) / TB, TB, 0, stream>>>(ei, sa, sb, bd, out, E);
}
